// Attention_20882130993855
// MI455X (gfx1250) — matrix operand order repaired and checked by reading, not compiled, not run on hardware
//
#include <hip/hip_runtime.h>

#define NUM_HEADS 16
#define NUM_KV    4
#define HDIM      128
#define QK_SCALE_LOG2E 0.12751744f   // (1/sqrt(128)) * log2(e); exp done as exp2
#define NEG_INF   -1.0e10f

typedef __attribute__((ext_vector_type(16))) _Float16 v16h;
typedef __attribute__((ext_vector_type(8)))  _Float16 v8h;
typedef __attribute__((ext_vector_type(2)))  _Float16 v2h;
typedef __attribute__((ext_vector_type(8)))  float    v8f;
typedef __attribute__((ext_vector_type(4)))  float    v4f;

#define KL_STRIDE 136  // halves per K row (128 dims + 8 pad) -> b128-aligned, bank-spread
#define VT_STRIDE 40   // halves per VT row (32 keys + 8 pad)
#define P_STRIDE  40   // halves per P row  (32 cols + 8 pad)

// wave32 XOR-lane reduction step via ds_swizzle (group-of-32: offset = 0x1F | xor<<10)
#define SWIZ_MAX(t, imm) \
  t = fmaxf(t, __int_as_float(__builtin_amdgcn_ds_swizzle(__float_as_int(t), (imm))))

__global__ __launch_bounds__(128) void fa_varlen_kernel(
    const float* __restrict__ q, const float* __restrict__ k,
    const float* __restrict__ v, const int* __restrict__ cu,
    float* __restrict__ out)
{
  __shared__ __align__(16) _Float16 Kl[32 * KL_STRIDE];     // K tile, f16 row-major [key][dim]
  __shared__ __align__(16) _Float16 VT[HDIM * VT_STRIDE];   // V tile, f16 transposed [dim][key]
  __shared__ __align__(16) _Float16 Pb[4][16 * P_STRIDE];   // per-wave P staging

  const int tid  = threadIdx.x;
  const int lane = tid & 31;
  const int wave = tid >> 5;
  const int lrow = lane & 15;   // row (A) or column (B/C) index
  const int lhi  = lane >> 4;   // lane-half selector

  const int h   = blockIdx.y;
  const int kvh = h >> 2;       // GQA: head h reads kv head h/4

  const int blk_row0 = blockIdx.x * 64;
  int seq = 0;
  #pragma unroll
  for (int i = 1; i < 8; ++i) if (blk_row0 >= cu[i]) seq = i;
  const int seq_start = cu[seq];
  const int seq_len   = cu[seq + 1] - seq_start;

  const int row0   = blk_row0 + wave * 16;         // this wave's first token row
  const int m0     = row0 - seq_start;             // wave-local query offset
  const int kv_lim = (blk_row0 - seq_start) + 64;  // keys needed by whole block (causal)

  // ---------------- Q tile -> 4 WMMA A-fragments (K chunks of 32 dims) --------------
  const float* qrow = q + (size_t)(row0 + lrow) * (NUM_HEADS * HDIM) + h * HDIM;
  v16h Aq[4];
  #pragma unroll
  for (int c = 0; c < 4; ++c) {
    const float* p = qrow + c * 32 + lhi * 8;
    v4f x0 = *(const v4f*)(p);
    v4f x1 = *(const v4f*)(p + 4);
    v4f y0 = *(const v4f*)(p + 16);
    v4f y1 = *(const v4f*)(p + 20);
    #pragma unroll
    for (int i = 0; i < 4; ++i) {
      Aq[c][i]      = (_Float16)x0[i];
      Aq[c][4 + i]  = (_Float16)x1[i];
      Aq[c][8 + i]  = (_Float16)y0[i];
      Aq[c][12 + i] = (_Float16)y1[i];
    }
  }

  // ones B-fragment for row-sum-by-WMMA
  v16h onesB;
  #pragma unroll
  for (int i = 0; i < 16; ++i) onesB[i] = (_Float16)1.0f;

  v8f O[8];
  #pragma unroll
  for (int d = 0; d < 8; ++d) O[d] = (v8f)0.0f;
  float mrun[8], lrun[8];
  #pragma unroll
  for (int r = 0; r < 8; ++r) { mrun[r] = -3.0e38f; lrun[r] = 0.0f; }

  _Float16* Pw = &Pb[wave][0];

  for (int n0 = 0; n0 < kv_lim; n0 += 32) {
    // ---- cooperative staging: K (row-major f16) + V (transposed f16), 128 threads --
    __syncthreads();   // protect Kl/VT from previous iteration's readers
    {
      // K: thread owns (key = tid&31, 32-dim chunk = tid>>5); b128 LDS stores
      const int kcol  = tid & 31;
      const int dbase = (tid >> 5) * 32;
      int keyl = n0 + kcol;
      int kc   = keyl < seq_len ? keyl : seq_len - 1;   // clamp (masked / P==0 anyway)
      const size_t rowoff = (size_t)(seq_start + kc) * (NUM_KV * HDIM) + kvh * HDIM + dbase;
      const float* kg = k + rowoff;
      #pragma unroll
      for (int dd = 0; dd < 32; dd += 8) {
        v4f a0 = *(const v4f*)(kg + dd);
        v4f a1 = *(const v4f*)(kg + dd + 4);
        v8h hh;
        #pragma unroll
        for (int i = 0; i < 4; ++i) { hh[i] = (_Float16)a0[i]; hh[4 + i] = (_Float16)a1[i]; }
        *(v8h*)&Kl[kcol * KL_STRIDE + dbase + dd] = hh;
      }
      // V: thread owns (key pair = tid&15, 16-dim chunk = tid>>4); packed b32 stores
      const int kp2 = tid & 15;
      const int dch = tid >> 4;
      int ke = n0 + 2 * kp2;
      int kce = ke     < seq_len ? ke     : seq_len - 1;
      int kco = ke + 1 < seq_len ? ke + 1 : seq_len - 1;
      const float* vg0 = v + (size_t)(seq_start + kce) * (NUM_KV * HDIM) + kvh * HDIM + dch * 16;
      const float* vg1 = v + (size_t)(seq_start + kco) * (NUM_KV * HDIM) + kvh * HDIM + dch * 16;
      #pragma unroll
      for (int dd = 0; dd < 16; dd += 4) {
        v4f a = *(const v4f*)(vg0 + dd);
        v4f b = *(const v4f*)(vg1 + dd);
        #pragma unroll
        for (int i = 0; i < 4; ++i) {
          v2h pr; pr[0] = (_Float16)a[i]; pr[1] = (_Float16)b[i];
          *(v2h*)&VT[(dch * 16 + dd + i) * VT_STRIDE + 2 * kp2] = pr;
        }
      }
      // prefetch next iteration's K/V rows into cache (global_prefetch_b8)
      if (n0 + 32 < kv_lim) {
        int keyn = keyl + 32;
        if (keyn >= seq_len) keyn = seq_len - 1;
        const size_t noff = (size_t)(seq_start + keyn) * (NUM_KV * HDIM) + kvh * HDIM + dbase;
        __builtin_prefetch((const void*)(k + noff), 0, 3);
        __builtin_prefetch((const void*)(v + noff), 0, 3);
      }
    }
    __syncthreads();

    // ---- per-wave compute; skip blocks fully above the causal diagonal -------------
    if (n0 <= m0 + 15) {
      // scores S(16x32) = Q @ K^T : 2 column tiles x 4 chained WMMAs, K from LDS
      v8f S[2];
      #pragma unroll
      for (int j = 0; j < 2; ++j) {
        v8f acc = (v8f)0.0f;
        #pragma unroll
        for (int c = 0; c < 4; ++c) {
          const _Float16* kp = &Kl[(j * 16 + lrow) * KL_STRIDE + c * 32 + lhi * 8];
          v8h b0 = *(const v8h*)(kp);
          v8h b1 = *(const v8h*)(kp + 16);
          v16h Bk;
          #pragma unroll
          for (int i = 0; i < 8; ++i) { Bk[i] = b0[i]; Bk[8 + i] = b1[i]; }
          acc = __builtin_amdgcn_wmma_f32_16x16x32_f16(
                    false, Aq[c], false, Bk, (short)0, acc, false, false);
        }
        S[j] = acc;
      }

      // online softmax (log2 domain): row max via ds_swizzle XOR reductions
      const bool needmask = (n0 + 31) > m0;   // wave-uniform
      const int key0 = n0 + lrow;
      const int key1 = key0 + 16;
      float px0[8], px1[8], alphav[8];
      bool changed = false;
      #pragma unroll
      for (int r = 0; r < 8; ++r) {
        int qpos = m0 + lhi * 8 + r;
        float s0 = S[0][r] * QK_SCALE_LOG2E;
        float s1 = S[1][r] * QK_SCALE_LOG2E;
        if (needmask) {
          if (key0 > qpos) s0 = NEG_INF;
          if (key1 > qpos) s1 = NEG_INF;
        }
        float t = fmaxf(s0, s1);
        SWIZ_MAX(t, 0x041F);   // xor 1
        SWIZ_MAX(t, 0x081F);   // xor 2
        SWIZ_MAX(t, 0x101F);   // xor 4
        SWIZ_MAX(t, 0x201F);   // xor 8
        float mo = mrun[r];
        float mn = fmaxf(mo, t);
        changed |= (mn != mo);
        alphav[r] = exp2f(mo - mn);
        px0[r]    = exp2f(s0 - mn);
        px1[r]    = exp2f(s1 - mn);
        mrun[r]   = mn;
      }
      // rescale O only if some row's max moved (ballot is wave-uniform)
      if (__ballot(changed) != 0ull) {
        #pragma unroll
        for (int r = 0; r < 8; ++r) {
          float a = alphav[r];
          #pragma unroll
          for (int d = 0; d < 8; ++d) O[d][r] *= a;
        }
      }

      // P (C layout) -> LDS -> A-fragment (wave-private round-trip)
      #pragma unroll
      for (int r = 0; r < 8; ++r) {
        Pw[(lhi * 8 + r) * P_STRIDE + lrow]      = (_Float16)px0[r];
        Pw[(lhi * 8 + r) * P_STRIDE + 16 + lrow] = (_Float16)px1[r];
      }
      asm volatile("s_wait_dscnt 0" ::: "memory");

      v16h Pa;
      {
        v8h p0 = *(const v8h*)(Pw + lrow * P_STRIDE + lhi * 8);
        v8h p1 = *(const v8h*)(Pw + lrow * P_STRIDE + 16 + lhi * 8);
        #pragma unroll
        for (int i = 0; i < 8; ++i) { Pa[i] = p0[i]; Pa[8 + i] = p1[i]; }
      }

      // row sums by WMMA: P @ ones -> every column of D holds the row sum
      v8f rs = __builtin_amdgcn_wmma_f32_16x16x32_f16(
                   false, Pa, false, onesB, (short)0, (v8f)0.0f, false, false);
      #pragma unroll
      for (int r = 0; r < 8; ++r) lrun[r] = lrun[r] * alphav[r] + rs[r];

      // O(16x128) += P(16x32) @ V(32x128) : 8 WMMAs fed from transposed V in LDS
      #pragma unroll
      for (int dt = 0; dt < 8; ++dt) {
        const _Float16* vp = &VT[(dt * 16 + lrow) * VT_STRIDE + lhi * 8];
        v8h b0 = *(const v8h*)(vp);
        v8h b1 = *(const v8h*)(vp + 16);
        v16h Bv;
        #pragma unroll
        for (int i = 0; i < 8; ++i) { Bv[i] = b0[i]; Bv[8 + i] = b1[i]; }
        O[dt] = __builtin_amdgcn_wmma_f32_16x16x32_f16(
                    false, Pa, false, Bv, (short)0, O[dt], false, false);
      }
    }
  }

  // ---------------- epilogue: O * (1/l) via v_rcp, scatter to ragged output ---------
  float linv[8];
  #pragma unroll
  for (int r = 0; r < 8; ++r) linv[r] = __builtin_amdgcn_rcpf(lrun[r]);
  #pragma unroll
  for (int dt = 0; dt < 8; ++dt) {
    #pragma unroll
    for (int r = 0; r < 8; ++r) {
      int row = lhi * 8 + r;
      out[(size_t)(row0 + row) * (NUM_HEADS * HDIM) + h * HDIM + dt * 16 + lrow]
          = O[dt][r] * linv[r];
    }
  }
}

extern "C" void kernel_launch(void* const* d_in, const int* in_sizes, int n_in,
                              void* d_out, int out_size, void* d_ws, size_t ws_size,
                              hipStream_t stream) {
  (void)n_in; (void)out_size; (void)d_ws; (void)ws_size;
  const float* q  = (const float*)d_in[0];
  const float* k  = (const float*)d_in[1];
  const float* v  = (const float*)d_in[2];
  const int*   cu = (const int*)d_in[3];
  float* out = (float*)d_out;

  const int total = in_sizes[0] / (NUM_HEADS * HDIM);   // 6528 ragged tokens
  dim3 grid(total / 64, NUM_HEADS);                     // 64-row M blocks (lens % 64 == 0)
  dim3 block(128);                                      // 4 waves, 16 query rows each
  fa_varlen_kernel<<<grid, block, 0, stream>>>(q, k, v, cu, out);
}
